// Policy_88905823027287
// MI455X (gfx1250) — compile-verified, matrix operand order checked by reading, not run on hardware
//
#include <hip/hip_runtime.h>
#include <hip/hip_bf16.h>
#include <math.h>

// ---------------- problem constants ----------------
#define NQ      20
#define DIM     (1 << NQ)          // 1048576 amplitudes
#define LAYERS  3
#define NGATES  300                // 3 layers * 5 sweeps * 20 qubits

// ---------------- tiling constants -----------------
#define CL      13                 // chunk bits (8192 amplitudes = 64 KB LDS)
#define CHUNK   (1 << CL)
#define TPB     256

typedef __attribute__((ext_vector_type(16))) _Float16 v16h;
typedef __attribute__((ext_vector_type(8)))  _Float16 v8h;
typedef __attribute__((ext_vector_type(8)))  float    v8f;
typedef __attribute__((ext_vector_type(4)))  float    v4f;
typedef __attribute__((ext_vector_type(4)))  int      v4i_;

// CDNA5 async global<->LDS copy paths (guarded: fall back cleanly if the
// toolchain doesn't declare the builtins).
#if defined(__gfx1250__) && __has_builtin(__builtin_amdgcn_global_load_async_to_lds_b128) && __has_builtin(__builtin_amdgcn_s_wait_asynccnt)
#define HAVE_ASYNC_LDS 1
typedef __attribute__((address_space(1))) v4i_* gptr_b128;
typedef __attribute__((address_space(3))) v4i_* lptr_b128;
__device__ __forceinline__ void async_copy_b128(const void* gsrc, void* ldst) {
    __builtin_amdgcn_global_load_async_to_lds_b128(
        (gptr_b128)(void*)gsrc, (lptr_b128)ldst, 0, 0);
}
#else
#define HAVE_ASYNC_LDS 0
#endif

#if HAVE_ASYNC_LDS && __has_builtin(__builtin_amdgcn_global_store_async_from_lds_b128)
#define HAVE_ASYNC_ST 1
__device__ __forceinline__ void async_store_b128(void* gdst, const void* lsrc) {
    __builtin_amdgcn_global_store_async_from_lds_b128(
        (gptr_b128)gdst, (lptr_b128)(void*)lsrc, 0, 0);
}
#else
#define HAVE_ASYNC_ST 0
#endif

// =====================================================================
// 1) Build all 300 ZYZ gate matrices from (x, weight0, theta).
//    p[i] = weight0[i] * x[i % 14] + theta[i]
//    Gate g: layer k=g/100, sweep s=(g%100)/20, qubit q=g%20.
// =====================================================================
__global__ void qsim_prep_gates(const float* __restrict__ x,
                                const float* __restrict__ w0,
                                const float* __restrict__ th,
                                float* __restrict__ gates) {
    int g = blockIdx.x * blockDim.x + threadIdx.x;
    if (g >= NGATES) return;
    int k = g / 100, r = g % 100, s = r / 20, q = r % 20;
    float a, b, c;
    if (s < 4) {
        int off = k * 840 + s * 60 + q * 3;
        a = w0[off]     * x[off % 14]       + th[off];
        b = w0[off + 1] * x[(off + 1) % 14] + th[off + 1];
        c = w0[off + 2] * x[(off + 2) % 14] + th[off + 2];
    } else {
        int off = k * 840 + 240 + q * 2;
        a = w0[off]     * x[off % 14]       + th[off];
        b = w0[off + 1] * x[(off + 1) % 14] + th[off + 1];
        c = 0.0f;
    }
    float cb = cosf(0.5f * b), sb = sinf(0.5f * b);
    float apc = 0.5f * (a + c), amc = 0.5f * (a - c);
    float epr = cosf(apc), epi = -sinf(apc);   // e^{-i(a+c)/2}
    float emr = cosf(amc), emi =  sinf(amc);   // e^{ i(a-c)/2}
    float* o = gates + (size_t)g * 8;
    o[0] =  epr * cb;  o[1] =  epi * cb;       // U00 = ep * cb
    o[2] = -emr * sb;  o[3] = -emi * sb;       // U01 = -em * sb
    o[4] =  emr * sb;  o[5] = -emi * sb;       // U10 = conj(em) * sb
    o[6] =  epr * cb;  o[7] = -epi * cb;       // U11 = conj(ep) * cb
}

// =====================================================================
// 2) |0...0> initial state + zero partial-logit accumulator.
// =====================================================================
__global__ void qsim_init(float2* __restrict__ st, float* __restrict__ partial) {
    int i = blockIdx.x * blockDim.x + threadIdx.x;
    if (i < DIM) st[i] = make_float2(i == 0 ? 1.0f : 0.0f, 0.0f);
    if (i < 16)  partial[i] = 0.0f;
}

// complex 2x2 butterfly on an LDS chunk for local bit `bit`
__device__ __forceinline__ void butterfly(float2* s, const float* g, int bit) {
    const float u00r = g[0], u00i = g[1], u01r = g[2], u01i = g[3];
    const float u10r = g[4], u10i = g[5], u11r = g[6], u11i = g[7];
    const int mask = (1 << bit) - 1;
    for (int t = threadIdx.x; t < (CHUNK >> 1); t += TPB) {
        int i0 = ((t & ~mask) << 1) | (t & mask);
        int i1 = i0 | (1 << bit);
        float2 av = s[i0], bv = s[i1];
        float2 r0, r1;
        r0.x = u00r * av.x - u00i * av.y + u01r * bv.x - u01i * bv.y;
        r0.y = u00r * av.y + u00i * av.x + u01r * bv.y + u01i * bv.x;
        r1.x = u10r * av.x - u10i * av.y + u11r * bv.x - u11i * bv.y;
        r1.y = u10r * av.y + u10i * av.x + u11r * bv.y + u11i * bv.x;
        s[i0] = r0; s[i1] = r1;
    }
}

// =====================================================================
// 3a) Sweep over flat bits 0..12 (contiguous 8K chunk in 64 KB LDS).
//     Reference qubit for flat bit bp is (19 - bp).
// =====================================================================
__global__ void __launch_bounds__(TPB)
qsim_sweep_low(float2* __restrict__ st, const float* __restrict__ gates, int sweepBase) {
    __shared__ float2 s[CHUNK];
    const int base = blockIdx.x << CL;
    __builtin_prefetch(st + base, 0, 3);
#if HAVE_ASYNC_LDS
    for (int i = threadIdx.x * 2; i < CHUNK; i += TPB * 2)
        async_copy_b128(st + base + i, &s[i]);
    __builtin_amdgcn_s_wait_asynccnt(0);
    __syncthreads();
#else
    for (int i = threadIdx.x; i < CHUNK; i += TPB) s[i] = st[base + i];
    __syncthreads();
#endif
    for (int bp = 0; bp < CL; ++bp) {                   // flat bits 0..12
        const float* g = gates + (size_t)(sweepBase + (19 - bp)) * 8;
        butterfly(s, g, bp);
        __syncthreads();
    }
#if HAVE_ASYNC_ST
    for (int i = threadIdx.x * 2; i < CHUNK; i += TPB * 2)
        async_store_b128(st + base + i, &s[i]);
    __builtin_amdgcn_s_wait_asynccnt(0);
#else
    for (int i = threadIdx.x; i < CHUNK; i += TPB) st[base + i] = s[i];
#endif
}

// =====================================================================
// 3b) Sweep over flat bits 13..19. Chunk covers bits {0..5} U {13..19};
//     blockIdx supplies bits 6..12. Optionally fuses the CZ ring sign
//     (applied in LDS so the DMA-out path serves both cases).
// =====================================================================
__global__ void __launch_bounds__(TPB)
qsim_sweep_high(float2* __restrict__ st, const float* __restrict__ gates,
                int sweepBase, int doCZ) {
    __shared__ float2 s[CHUNK];
    const int mid = blockIdx.x;                         // bits 6..12 (7 bits)
#if HAVE_ASYNC_LDS
    // even i: elements i, i+1 are globally contiguous (lo6 < 63) -> b128 legal
    for (int i = threadIdx.x * 2; i < CHUNK; i += TPB * 2) {
        int g = (i & 63) | (mid << 6) | ((i >> 6) << 13);
        async_copy_b128(st + g, &s[i]);
    }
    __builtin_amdgcn_s_wait_asynccnt(0);
    __syncthreads();
#else
    for (int i = threadIdx.x; i < CHUNK; i += TPB) {
        int g = (i & 63) | (mid << 6) | ((i >> 6) << 13);
        s[i] = st[g];
    }
    __syncthreads();
#endif
    for (int t = 0; t < 7; ++t) {                       // flat bit 13+t -> local bit 6+t
        const float* g = gates + (size_t)(sweepBase + (19 - (13 + t))) * 8;
        butterfly(s, g, 6 + t);
        __syncthreads();
    }
    if (doCZ) {                                         // CZ ring sign, in LDS
        for (int i = threadIdx.x; i < CHUNK; i += TPB) {
            unsigned g = (unsigned)((i & 63) | (mid << 6) | ((i >> 6) << 13));
            unsigned pairs = (g & (g >> 1)) & 0x7FFFFu;           // pairs (j,j+1), j=0..18
            int par = (__popc(pairs) + (int)((g >> 19) & g & 1u)) & 1;
            float sgn = par ? -1.0f : 1.0f;
            float2 v = s[i];
            v.x *= sgn; v.y *= sgn;
            s[i] = v;
        }
        __syncthreads();
    }
#if HAVE_ASYNC_ST
    for (int i = threadIdx.x * 2; i < CHUNK; i += TPB * 2) {
        int g = (i & 63) | (mid << 6) | ((i >> 6) << 13);
        async_store_b128(st + g, &s[i]);
    }
    __builtin_amdgcn_s_wait_asynccnt(0);
#else
    for (int i = threadIdx.x; i < CHUNK; i += TPB) {
        int g = (i & 63) | (mid << 6) | ((i >> 6) << 13);
        st[g] = s[i];
    }
#endif
}

// =====================================================================
// 4) probs[i] = |amp|^2, stored as f16 for the WMMA GEMV.
// =====================================================================
__global__ void qsim_probs(const float2* __restrict__ st, _Float16* __restrict__ ph) {
    int i = blockIdx.x * blockDim.x + threadIdx.x;
    float2 v = st[i];
    ph[i] = (_Float16)(v.x * v.x + v.y * v.y);
}

// =====================================================================
// 5) logits = q1_w @ probs via v_wmma_f32_16x16x32_f16.
//    Per-lane A/B halves are two contiguous 8-k runs (k0..k0+7, k0+16..k0+23,
//    k0 = base + 8*hi) -> unconditional b128 loads + value selects (no EXEC
//    branching). Two accumulators break the C dependency chain.
// =====================================================================
#define KPW (1 << 12)   // 4096 k per wave -> 256 waves total

__global__ void __launch_bounds__(256)
qsim_logits_wmma(const float* __restrict__ qw, const _Float16* __restrict__ ph,
                 float* __restrict__ partial) {
    const int wave  = (blockIdx.x * blockDim.x + threadIdx.x) >> 5;
    const int lane  = threadIdx.x & 31;
    const int m     = lane & 15;                 // A row / B col
    const int hi    = lane >> 4;                 // K half-group select
    const bool arow = (m < 14);
    const bool bcol = (m == 0);
    const float* rowp = qw + (size_t)(arow ? m : 0) * DIM;  // clamp; zeroed below
    const int kbase = wave * KPW;
    const v16h zerov = {};

    v8f c0 = {}, c1 = {};
    for (int kb = 0; kb < KPW; kb += 64) {
        #pragma unroll
        for (int u = 0; u < 2; ++u) {
            const int k0 = kbase + kb + 32 * u + 8 * hi;
            // ---- A: q1_w row, four b128 loads + pack-convert to f16 ----
            v4f w0 = *(const v4f*)(rowp + k0);
            v4f w1 = *(const v4f*)(rowp + k0 + 4);
            v4f w2 = *(const v4f*)(rowp + k0 + 16);
            v4f w3 = *(const v4f*)(rowp + k0 + 20);
            v16h a;
            #pragma unroll
            for (int j = 0; j < 4; ++j) {
                a[j]      = (_Float16)w0[j];
                a[4 + j]  = (_Float16)w1[j];
                a[8 + j]  = (_Float16)w2[j];
                a[12 + j] = (_Float16)w3[j];
            }
            a = arow ? a : zerov;                // rows 14/15 -> 0 (cndmask)
            // ---- B: probs, two b128 loads, nonzero only in column 0 ----
            v8h p0 = *(const v8h*)(ph + k0);
            v8h p1 = *(const v8h*)(ph + k0 + 16);
            v16h b;
            #pragma unroll
            for (int j = 0; j < 8; ++j) { b[j] = p0[j]; b[8 + j] = p1[j]; }
            b = bcol ? b : zerov;                // cols 1..15 -> 0 (cndmask)
            if (u == 0)
                c0 = __builtin_amdgcn_wmma_f32_16x16x32_f16(false, a, false, b,
                                                            (short)0, c0, false, false);
            else
                c1 = __builtin_amdgcn_wmma_f32_16x16x32_f16(false, a, false, b,
                                                            (short)0, c1, false, false);
        }
    }
    // D layout: lane 0 holds N=0 rows 0..7 in c[0..7]; lane 16 rows 8..15.
    if ((lane & 15) == 0) {
        #pragma unroll
        for (int v = 0; v < 8; ++v) {
            int row = v + 8 * hi;
            if (row < 14) atomicAdd(&partial[row], c0[v] + c1[v]);
        }
    }
}

// =====================================================================
// 6) bias + softmax over 14 logits (tiny, single thread).
// =====================================================================
__global__ void qsim_softmax(const float* __restrict__ partial,
                             const float* __restrict__ bias,
                             float* __restrict__ out) {
    if (threadIdx.x != 0 || blockIdx.x != 0) return;
    float l[14], mx = -3.0e38f;
    for (int j = 0; j < 14; ++j) { l[j] = partial[j] + bias[j]; mx = fmaxf(mx, l[j]); }
    float sum = 0.0f;
    for (int j = 0; j < 14; ++j) { l[j] = expf(l[j] - mx); sum += l[j]; }
    for (int j = 0; j < 14; ++j) out[j] = l[j] / sum;
}

// =====================================================================
// launcher
// =====================================================================
extern "C" void kernel_launch(void* const* d_in, const int* in_sizes, int n_in,
                              void* d_out, int out_size, void* d_ws, size_t ws_size,
                              hipStream_t stream) {
    (void)in_sizes; (void)n_in; (void)out_size; (void)ws_size;
    const float* x    = (const float*)d_in[0];
    const float* w0   = (const float*)d_in[1];
    const float* th   = (const float*)d_in[2];
    const float* qw   = (const float*)d_in[3];
    const float* q1b  = (const float*)d_in[4];
    float*       out  = (float*)d_out;

    // workspace layout
    char* ws = (char*)d_ws;
    float2*   st      = (float2*)ws;                                   // 8 MB
    float*    gates   = (float*)(ws + (size_t)DIM * 8);                // 9.6 KB
    _Float16* ph      = (_Float16*)(ws + (size_t)DIM * 8 + 16384);     // 2 MB
    float*    partial = (float*)(ws + (size_t)DIM * 8 + 16384 + (size_t)DIM * 2);

    qsim_prep_gates<<<1, 320, 0, stream>>>(x, w0, th, gates);
    qsim_init<<<DIM / 256, 256, 0, stream>>>(st, partial);

    for (int k = 0; k < LAYERS; ++k) {
        for (int s = 0; s < 5; ++s) {
            int gb   = (k * 5 + s) * 20;
            int doCZ = (s == 4 && k != LAYERS - 1) ? 1 : 0;
            qsim_sweep_low <<<DIM / CHUNK, TPB, 0, stream>>>(st, gates, gb);
            qsim_sweep_high<<<1 << (NQ - CL), TPB, 0, stream>>>(st, gates, gb, doCZ);
        }
    }

    qsim_probs<<<DIM / 256, 256, 0, stream>>>(st, ph);
    qsim_logits_wmma<<<(DIM / KPW) * 32 / 256, 256, 0, stream>>>(qw, ph, partial);
    qsim_softmax<<<1, 32, 0, stream>>>(partial, q1b, out);
}
